// GPTSelfAttention1D_22488448762289
// MI455X (gfx1250) — matrix operand order repaired and checked by reading, not compiled, not run on hardware
//
#include <hip/hip_runtime.h>
#include <hip/hip_bf16.h>

typedef __attribute__((ext_vector_type(16))) _Float16 v16h;
typedef __attribute__((ext_vector_type(8)))  _Float16 v8h;
typedef __attribute__((ext_vector_type(8)))  float    v8f;

#define DIMD   1024
#define NHEADS 16
#define HDIM   64
#define BATCH  2
#define SEQ    2048
#define NTOK   (BATCH * SEQ)   // 4096
#define WPB    8               // waves per block (256 threads, wave32)

#define WMMA_F16(a, b, c) \
  __builtin_amdgcn_wmma_f32_16x16x32_f16(false, (a), false, (b), (short)0, (c), false, false)

// ---------- fragment loaders (global memory, f16) ----------
// A-fragment 16x32 (MxK): lane m=lane&15 holds row m.
// lanes 0-15: K = {0..7, 16..23}; lanes 16-31: K = {8..15, 24..31}.
static __device__ __forceinline__ v16h ld_afrag(const _Float16* base, int ldm, int lane) {
  const int m  = lane & 15;
  const int kb = (lane < 16) ? 0 : 8;
  const _Float16* p = base + (size_t)m * ldm + kb;
  v8h lo = *(const v8h*)(p);
  v8h hi = *(const v8h*)(p + 16);
  v16h a;
#pragma unroll
  for (int j = 0; j < 8; ++j) { a[j] = lo[j]; a[j + 8] = hi[j]; }
  return a;
}

// B-fragment 32x16 (KxN): lane n=lane&15 holds column n.
// lanes 0-15: K = {0..7, 16..23}; lanes 16-31: K = {8..15, 24..31}.
// colbase points to column n's 32 contiguous K values.
static __device__ __forceinline__ v16h ld_bfrag(const _Float16* colbase, int lane) {
  const int kb = (lane < 16) ? 0 : 8;
  v8h lo = *(const v8h*)(colbase + kb);
  v8h hi = *(const v8h*)(colbase + kb + 16);
  v16h b;
#pragma unroll
  for (int j = 0; j < 8; ++j) { b[j] = lo[j]; b[j + 8] = hi[j]; }
  return b;
}

// ---------- conversion kernels ----------
__global__ void cvt_f32_to_f16(const float* __restrict__ in, _Float16* __restrict__ out, int n) {
  int i = blockIdx.x * blockDim.x + threadIdx.x;
  if (i < n) out[i] = (_Float16)in[i];
}

// Wt[n*K + k] = (f16) W[k*N + n]
__global__ void cvt_transpose_f16(const float* __restrict__ W, _Float16* __restrict__ Wt,
                                  int K, int N) {
  int i = blockIdx.x * blockDim.x + threadIdx.x;
  if (i >= K * N) return;
  int k = i / N, n = i % N;
  Wt[(size_t)n * K + k] = (_Float16)W[i];
}

// ---------- WMMA GEMM: out = A[M,K] x W[K,N] + bias ----------
// Wt is W transposed [N,K]. One wave -> 32x64 output tile (8 C-tiles).
// mode: 0 = f16 row-major [M,N]; 1 = f16 head-transposed V layout [B][H][hd][S];
//       2 = f32 row-major [M,N]
__global__ __launch_bounds__(256) void tile_gemm(
    const _Float16* __restrict__ A, const _Float16* __restrict__ Wt,
    const float* __restrict__ bias, _Float16* __restrict__ outh,
    float* __restrict__ outf, int M, int N, int K, int mode)
{
  const int lane  = threadIdx.x & 31;
  const int wid   = blockIdx.x * WPB + (threadIdx.x >> 5);
  const int nt64  = N >> 6;
  const int mt    = wid / nt64;
  const int nt    = wid % nt64;
  if (mt * 32 >= M) return;
  const int row0  = mt * 32;
  const int col0  = nt * 64;
  const int lanen = lane & 15;
  const int rb    = (lane < 16) ? 0 : 8;

  v8f c[2][4] = {};
  for (int k0 = 0; k0 < K; k0 += 32) {
    const v16h a0 = ld_afrag(A + (size_t)row0 * K + k0, K, lane);
    const v16h a1 = ld_afrag(A + (size_t)(row0 + 16) * K + k0, K, lane);
#pragma unroll
    for (int tt = 0; tt < 4; ++tt) {
      const v16h b = ld_bfrag(Wt + (size_t)(col0 + tt * 16 + lanen) * K + k0, lane);
      c[0][tt] = WMMA_F16(a0, b, c[0][tt]);
      c[1][tt] = WMMA_F16(a1, b, c[1][tt]);
    }
  }
#pragma unroll
  for (int mh = 0; mh < 2; ++mh) {
#pragma unroll
    for (int tt = 0; tt < 4; ++tt) {
      const int col = col0 + tt * 16 + lanen;
      const float bval = bias[col];
#pragma unroll
      for (int i = 0; i < 8; ++i) {
        const int row = row0 + mh * 16 + rb + i;
        const float val = c[mh][tt][i] + bval;
        if (mode == 0) {
          outh[(size_t)row * N + col] = (_Float16)val;
        } else if (mode == 1) {
          // V layout: [B][H][hd][S]
          const int bb = row >> 11, s = row & (SEQ - 1);
          const int h = col >> 6,  d = col & 63;
          outh[(((size_t)bb * NHEADS + h) * HDIM + d) * SEQ + s] = (_Float16)val;
        } else {
          outf[(size_t)row * N + col] = val;
        }
      }
    }
  }
}

// ---------- causal flash attention (one wave per b,h,16-query tile) ----------
// Vt is head-transposed: [B][H][hd][S]
__global__ __launch_bounds__(256) void attn_kernel(
    const _Float16* __restrict__ Qh, const _Float16* __restrict__ Kh,
    const _Float16* __restrict__ Vt, _Float16* __restrict__ ctxh)
{
  __shared__ _Float16 Plds[WPB][16][32];  // per-wave P staging (f16)
  const int lane  = threadIdx.x & 31;
  const int w     = threadIdx.x >> 5;
  const int wid   = blockIdx.x * WPB + w;
  const int qt    = wid & 127;          // S/16 = 128
  const int h     = (wid >> 7) & 15;
  const int b     = wid >> 11;
  const int lanen = lane & 15;
  const int rb    = (lane < 16) ? 0 : 8;

  const int    qpos0 = qt * 16;
  const size_t tok0  = (size_t)b * SEQ;
  const size_t qoff  = (tok0 + qpos0) * DIMD + h * HDIM;
  const size_t vhead = ((size_t)b * NHEADS + h) * HDIM * SEQ;  // base of [hd][S] slab

  const v16h aq0 = ld_afrag(Qh + qoff,      DIMD, lane);  // hd 0..31
  const v16h aq1 = ld_afrag(Qh + qoff + 32, DIMD, lane);  // hd 32..63

  // per-lane V column base pointers (column = head-dim coordinate)
  const _Float16* vcol0 = Vt + vhead + (size_t)( 0 + lanen) * SEQ;
  const _Float16* vcol1 = Vt + vhead + (size_t)(16 + lanen) * SEQ;
  const _Float16* vcol2 = Vt + vhead + (size_t)(32 + lanen) * SEQ;
  const _Float16* vcol3 = Vt + vhead + (size_t)(48 + lanen) * SEQ;

  float mi[8], li[8];
  v8f acc0 = {}, acc1 = {}, acc2 = {}, acc3 = {};
#pragma unroll
  for (int i = 0; i < 8; ++i) { mi[i] = -1e30f; li[i] = 0.0f; }

  for (int kk = 0; kk < qpos0 + 16; kk += 32) {
    // ---- scores S = (Q K^T)/8 : two 16x16 tiles ----
    v8f s0 = {}, s1 = {};
    {
      const _Float16* kb0 = Kh + (tok0 + kk + lanen) * DIMD + h * HDIM;
      s0 = WMMA_F16(aq0, ld_bfrag(kb0, lane), s0);
      s0 = WMMA_F16(aq1, ld_bfrag(kb0 + 32, lane), s0);
      const _Float16* kb1 = kb0 + (size_t)16 * DIMD;
      s1 = WMMA_F16(aq0, ld_bfrag(kb1, lane), s1);
      s1 = WMMA_F16(aq1, ld_bfrag(kb1 + 32, lane), s1);
    }
    // ---- scale + causal mask + running row max ----
    float tmax[8];
#pragma unroll
    for (int i = 0; i < 8; ++i) {
      const int q = qpos0 + rb + i;
      float v0 = s0[i] * 0.125f;             // 1/sqrt(64)
      float v1 = s1[i] * 0.125f;
      if (kk +      lanen > q) v0 = -10000.0f;
      if (kk + 16 + lanen > q) v1 = -10000.0f;
      s0[i] = v0; s1[i] = v1;
      tmax[i] = fmaxf(v0, v1);
    }
#pragma unroll
    for (int off = 1; off < 16; off <<= 1)
#pragma unroll
      for (int i = 0; i < 8; ++i)
        tmax[i] = fmaxf(tmax[i], __shfl_xor(tmax[i], off, 32));
    // ---- online softmax; stage P into LDS as f16 ----
    float sf[8], rs[8];
#pragma unroll
    for (int i = 0; i < 8; ++i) {
      const float mnew = fmaxf(mi[i], tmax[i]);
      sf[i] = __expf(mi[i] - mnew);
      mi[i] = mnew;
      const float p0 = __expf(s0[i] - mnew);
      const float p1 = __expf(s1[i] - mnew);
      rs[i] = p0 + p1;
      Plds[w][rb + i][lanen]      = (_Float16)p0;
      Plds[w][rb + i][16 + lanen] = (_Float16)p1;
    }
#pragma unroll
    for (int off = 1; off < 16; off <<= 1)
#pragma unroll
      for (int i = 0; i < 8; ++i)
        rs[i] += __shfl_xor(rs[i], off, 32);
#pragma unroll
    for (int i = 0; i < 8; ++i) {
      li[i] = li[i] * sf[i] + rs[i];
      acc0[i] *= sf[i]; acc1[i] *= sf[i]; acc2[i] *= sf[i]; acc3[i] *= sf[i];
    }
    asm volatile("s_wait_dscnt 0x0" ::: "memory");  // cross-lane LDS RAW within wave
    // ---- reload P as A-fragment (16x32) ----
    v16h ap;
    {
      const int kb = (lane < 16) ? 0 : 8;
#pragma unroll
      for (int j = 0; j < 8; ++j) {
        ap[j]     = Plds[w][lanen][kb + j];
        ap[j + 8] = Plds[w][lanen][16 + kb + j];
      }
    }
    asm volatile("" ::: "memory");
    // ---- ctx += P (16x32) x V (32x64) : contiguous b128 loads from Vt ----
    acc0 = WMMA_F16(ap, ld_bfrag(vcol0 + kk, lane), acc0);
    acc1 = WMMA_F16(ap, ld_bfrag(vcol1 + kk, lane), acc1);
    acc2 = WMMA_F16(ap, ld_bfrag(vcol2 + kk, lane), acc2);
    acc3 = WMMA_F16(ap, ld_bfrag(vcol3 + kk, lane), acc3);
  }
  // ---- normalize and write ctx (f16) ----
  float inv[8];
#pragma unroll
  for (int i = 0; i < 8; ++i) inv[i] = 1.0f / li[i];
  _Float16* cbase = ctxh + (tok0 + qpos0) * DIMD + h * HDIM;
#pragma unroll
  for (int i = 0; i < 8; ++i) {
    const size_t r = (size_t)(rb + i) * DIMD;
    cbase[r +  0 + lanen] = (_Float16)(acc0[i] * inv[i]);
    cbase[r + 16 + lanen] = (_Float16)(acc1[i] * inv[i]);
    cbase[r + 32 + lanen] = (_Float16)(acc2[i] * inv[i]);
    cbase[r + 48 + lanen] = (_Float16)(acc3[i] * inv[i]);
  }
}

extern "C" void kernel_launch(void* const* d_in, const int* in_sizes, int n_in,
                              void* d_out, int out_size, void* d_ws, size_t ws_size,
                              hipStream_t stream) {
  (void)in_sizes; (void)n_in; (void)out_size; (void)ws_size;
  const float* x  = (const float*)d_in[0];
  const float* Wq = (const float*)d_in[1];
  const float* bq = (const float*)d_in[2];
  const float* Wk = (const float*)d_in[3];
  const float* bk = (const float*)d_in[4];
  const float* Wv = (const float*)d_in[5];
  const float* bv = (const float*)d_in[6];
  const float* Wo = (const float*)d_in[7];
  const float* bo = (const float*)d_in[8];
  float* out = (float*)d_out;

  // workspace layout (f16 elements)
  _Float16* ws   = (_Float16*)d_ws;
  const size_t M4 = (size_t)NTOK * DIMD;       // 4M elements
  const size_t M1 = (size_t)DIMD * DIMD;       // 1M elements
  _Float16* xh   = ws;                         // 4M
  _Float16* WqT  = ws + M4;                    // 1M
  _Float16* WkT  = WqT + M1;
  _Float16* WvT  = WkT + M1;
  _Float16* WoT  = WvT + M1;
  _Float16* Qh   = WoT + M1;                   // 4M
  _Float16* Kh   = Qh + M4;                    // 4M
  _Float16* Vt   = Kh + M4;                    // 4M  (head-transposed V)
  _Float16* ctxh = Vt + M4;                    // 4M

  const int nx = NTOK * DIMD;
  cvt_f32_to_f16<<<(nx + 255) / 256, 256, 0, stream>>>(x, xh, nx);
  const int nw = DIMD * DIMD;
  cvt_transpose_f16<<<(nw + 255) / 256, 256, 0, stream>>>(Wq, WqT, DIMD, DIMD);
  cvt_transpose_f16<<<(nw + 255) / 256, 256, 0, stream>>>(Wk, WkT, DIMD, DIMD);
  cvt_transpose_f16<<<(nw + 255) / 256, 256, 0, stream>>>(Wv, WvT, DIMD, DIMD);
  cvt_transpose_f16<<<(nw + 255) / 256, 256, 0, stream>>>(Wo, WoT, DIMD, DIMD);

  // waves = (M/32)*(N/64) = 128*16 = 2048 -> 256 blocks of 8 waves
  const int gemm_blocks = (NTOK / 32) * (DIMD / 64) / WPB;
  tile_gemm<<<gemm_blocks, 256, 0, stream>>>(xh, WqT, bq, Qh, nullptr, NTOK, DIMD, DIMD, 0);
  tile_gemm<<<gemm_blocks, 256, 0, stream>>>(xh, WkT, bk, Kh, nullptr, NTOK, DIMD, DIMD, 0);
  tile_gemm<<<gemm_blocks, 256, 0, stream>>>(xh, WvT, bv, Vt, nullptr, NTOK, DIMD, DIMD, 1);

  // attention: B*H*(S/16) = 2*16*128 = 4096 waves -> 512 blocks
  attn_kernel<<<(BATCH * NHEADS * (SEQ / 16)) / WPB, 256, 0, stream>>>(Qh, Kh, Vt, ctxh);

  // output projection: ctx x Wo + bo -> f32
  tile_gemm<<<gemm_blocks, 256, 0, stream>>>(ctxh, WoT, bo, nullptr, out, NTOK, DIMD, DIMD, 2);
}
